// ImprovedRNAPocketEncoder_25022479466504
// MI455X (gfx1250) — matrix operand order repaired and checked by reading, not compiled, not run on hardware
//
#include <hip/hip_runtime.h>
#include <hip/hip_bf16.h>

// ---------------------------------------------------------------------------
// Problem constants (must match reference)
// ---------------------------------------------------------------------------
#define NE      160000      // edges
#define NN      5000        // nodes
#define NBATCH  8
#define MUL0    32
#define MUL1    16
#define MUL2    8
#define NW      144
#define RH      64
#define NBESS   8
#define NLAYER  4
#define POOLH   128
#define OUTD    512

typedef __attribute__((ext_vector_type(16))) _Float16 v16h;
typedef __attribute__((ext_vector_type(8)))  _Float16 v8h;
typedef __attribute__((ext_vector_type(8)))  float    v8f;

// fast activation helpers: v_rcp_f32 instead of IEEE divide chains
__device__ __forceinline__ float fast_rcp(float x) { return __builtin_amdgcn_rcpf(x); }
__device__ __forceinline__ float siluf(float x)    { return x * fast_rcp(1.0f + __expf(-x)); }
__device__ __forceinline__ float sigmf(float x)    { return fast_rcp(1.0f + __expf(-x)); }

// ---------------------------------------------------------------------------
// Workspace layout (float units)
// ---------------------------------------------------------------------------
constexpr size_t OFF_EMB  = 0;                      // NE*8 edge bessel*fcut
constexpr size_t OFF_Y    = OFF_EMB + (size_t)NE*8; // NE*8 : Y1(3) + Y2(5)
constexpr size_t OFF_S    = OFF_Y   + (size_t)NE*8; // NN*32
constexpr size_t OFF_V    = OFF_S   + (size_t)NN*32;  // NN*48
constexpr size_t OFF_Q    = OFF_V   + (size_t)NN*48;  // NN*40
constexpr size_t OFF_A0   = OFF_Q   + (size_t)NN*40;  // NN*56
constexpr size_t OFF_A1   = OFF_A0  + (size_t)NN*56;  // NN*144
constexpr size_t OFF_A2   = OFF_A1  + (size_t)NN*144; // NN*200
constexpr size_t OFF_LOG  = OFF_A2  + (size_t)NN*200; // NN
constexpr size_t OFF_GP   = OFF_LOG + (size_t)NN;     // 8*32
constexpr size_t OFF_PACK = OFF_GP  + 256;            // f16 packed weights start here
// Wr2 packed: 4 layers * 2 kb * 4 nt * 32 lanes * 16 halves = 16384 halves
// Wr3 packed: 4 layers * 2 kb * 9 nt * 32 lanes * 16 halves = 36864 halves

// ---------------------------------------------------------------------------
// Utility: zero a float range
// ---------------------------------------------------------------------------
__global__ void rna_zero_kernel(float* __restrict__ p, int n) {
    int i = blockIdx.x * 256 + threadIdx.x;
    if (i < n) p[i] = 0.0f;
}

// ---------------------------------------------------------------------------
// Pack Wr2 (NL x 64 x 64) into WMMA B-operand layout (f16):
// id = ((((layer*2+kb)*4+nt)*32+lane)*16 + h)
// lane holds column nt*16+(lane&15); halves are K = kb*32 + (h>>3)*16 + (lane>>4)*8 + (h&7)
// ---------------------------------------------------------------------------
__global__ void rna_pack_wr2_kernel(const float* __restrict__ Wr2, _Float16* __restrict__ W2p) {
    int id = blockIdx.x * 256 + threadIdx.x;
    if (id >= NLAYER * 4096) return;
    int h     = id & 15;
    int lane  = (id >> 4) & 31;
    int nt    = (id >> 9) & 3;
    int kb    = (id >> 11) & 1;
    int layer = id >> 12;
    int col = nt * 16 + (lane & 15);
    int K   = kb * 32 + (h >> 3) * 16 + (lane >> 4) * 8 + (h & 7);
    W2p[id] = (_Float16)Wr2[layer * 4096 + K * 64 + col];
}

// Wr3 (NL x 64 x 144), 9 N-tiles
__global__ void rna_pack_wr3_kernel(const float* __restrict__ Wr3, _Float16* __restrict__ W3p) {
    int id = blockIdx.x * 256 + threadIdx.x;
    if (id >= NLAYER * 9216) return;
    int h     = id & 15;
    int lane  = (id >> 4) & 31;
    int nt    = (id / 512) % 9;
    int kb    = (id / 4608) & 1;
    int layer = id / 9216;
    int col = nt * 16 + (lane & 15);
    int K   = kb * 32 + (h >> 3) * 16 + (lane >> 4) * 8 + (h & 7);
    W3p[id] = (_Float16)Wr3[layer * 9216 + K * 144 + col];
}

// ---------------------------------------------------------------------------
// Edge geometry: bessel*fcut embedding (8) and spherical harmonics Y1(3)+Y2(5)
// ---------------------------------------------------------------------------
__global__ void rna_geom_kernel(const float* __restrict__ pos, const int* __restrict__ ei,
                                float* __restrict__ emb, float* __restrict__ Yv) {
    int e = blockIdx.x * 256 + threadIdx.x;
    if (e >= NE) return;
    int si = ei[e], di = ei[NE + e];
    float vx = pos[si * 3 + 0] - pos[di * 3 + 0];
    float vy = pos[si * 3 + 1] - pos[di * 3 + 1];
    float vz = pos[si * 3 + 2] - pos[di * 3 + 2];
    float r2 = vx * vx + vy * vy + vz * vz + 1e-12f;
    float r  = sqrtf(r2);
    float ir = fast_rcp(r);
    float ux = vx * ir, uy = vy * ir, uz = vz * ir;
    float t  = fminf(r * (1.0f / 6.0f), 1.0f);
    float tt2 = t * t, tt3 = tt2 * t, t6 = tt3 * tt3, t7 = t6 * t, t8 = t7 * t;
    float fc = (1.0f - 28.0f * t6 + 48.0f * t7 - 21.0f * t8) * (r < 6.0f ? 1.0f : 0.0f);
    float pref = sqrtf(2.0f / 6.0f) * ir * fc;
    const float w0 = 3.14159265358979f / 6.0f;
    #pragma unroll
    for (int k = 0; k < NBESS; ++k)
        emb[(size_t)e * 8 + k] = pref * sinf((float)(k + 1) * w0 * r);
    const float s3 = 1.7320508075688772f, s15 = 3.872983346207417f, s5 = 2.23606797749979f;
    Yv[(size_t)e * 8 + 0] = s3 * ux;
    Yv[(size_t)e * 8 + 1] = s3 * uy;
    Yv[(size_t)e * 8 + 2] = s3 * uz;
    Yv[(size_t)e * 8 + 3] = s15 * ux * uy;
    Yv[(size_t)e * 8 + 4] = s15 * uy * uz;
    Yv[(size_t)e * 8 + 5] = 0.5f * s5 * (3.0f * uz * uz - 1.0f);
    Yv[(size_t)e * 8 + 6] = s15 * ux * uz;
    Yv[(size_t)e * 8 + 7] = 0.5f * s15 * (ux * ux - uy * uy);
}

// ---------------------------------------------------------------------------
// s = x @ W_in   (5000x64 @ 64x32)
// ---------------------------------------------------------------------------
__global__ void rna_init_s_kernel(const float* __restrict__ x, const float* __restrict__ Win,
                                  float* __restrict__ s) {
    int id = blockIdx.x * 256 + threadIdx.x;
    if (id >= NN * MUL0) return;
    int n = id >> 5, o = id & 31;
    float acc = 0.0f;
    #pragma unroll 8
    for (int c = 0; c < 64; ++c) acc += x[n * 64 + c] * Win[c * 32 + o];
    s[id] = acc;
}

// ---------------------------------------------------------------------------
// Fused edge kernel (one layer): radial MLP (WMMA) + message + scatter-add.
// Block = 256 threads (8 waves), tile = 128 edges, EXEC all-ones throughout.
// ---------------------------------------------------------------------------
__global__ __launch_bounds__(256)
void rna_edge_kernel(const int* __restrict__ ei,
                     const float* __restrict__ emb, const float* __restrict__ Yv,
                     const float* __restrict__ s, const float* __restrict__ v,
                     const float* __restrict__ q,
                     const float* __restrict__ Wr1, const float* __restrict__ br1,
                     const _Float16* __restrict__ W2p, const float* __restrict__ br2,
                     const _Float16* __restrict__ W3p, const float* __restrict__ br3,
                     float* __restrict__ a0, float* __restrict__ a1, float* __restrict__ a2) {
    __shared__ int   sSrc[128], sDst[128];
    __shared__ float sEmb[128][8];
    __shared__ float sY[128][8];
    __shared__ float sSe[128][32];
    __shared__ float sVe[128][48];
    __shared__ float sQe[128][40];
    __shared__ __align__(16) _Float16 sH1[128][64];
    __shared__ __align__(16) _Float16 sH2[128][64];

    const int tid  = threadIdx.x;
    const int e0   = blockIdx.x * 128;
    const int lane = tid & 31;
    const int wv   = tid >> 5;
    const int g    = lane >> 4;   // half-wave group
    const int ln16 = lane & 15;

    // ---------------- gather stage ----------------
    if (tid < 128) {
        sSrc[tid] = ei[e0 + tid];
        sDst[tid] = ei[NE + e0 + tid];
    }
    __syncthreads();
    for (int idx = tid; idx < 128 * 8; idx += 256) {
        int el = idx >> 3, k = idx & 7;
        sEmb[el][k] = emb[(size_t)(e0 + el) * 8 + k];
        sY[el][k]   = Yv[(size_t)(e0 + el) * 8 + k];
    }
    for (int idx = tid; idx < 128 * 32; idx += 256) {
        int el = idx >> 5, c = idx & 31;
        sSe[el][c] = s[sSrc[el] * 32 + c];
    }
    for (int idx = tid; idx < 128 * 48; idx += 256) {
        int el = idx / 48, c = idx % 48;
        sVe[el][c] = v[sSrc[el] * 48 + c];
    }
    for (int idx = tid; idx < 128 * 40; idx += 256) {
        int el = idx / 40, c = idx % 40;
        sQe[el][c] = q[sSrc[el] * 40 + c];
    }
    __syncthreads();

    // ---------------- stage 0: h1 = silu(emb @ Wr1 + br1)  (K=8, VALU) ------
    for (int idx = tid; idx < 128 * 64; idx += 256) {
        int el = idx >> 6, o = idx & 63;
        float acc = br1[o];
        #pragma unroll
        for (int k = 0; k < 8; ++k) acc += sEmb[el][k] * Wr1[k * 64 + o];
        sH1[el][o] = (_Float16)siluf(acc);
    }
    __syncthreads();

    // ---------------- stage 1: h2 = silu(h1 @ Wr2 + br2)  (WMMA) -----------
    // wave wv owns M-tile wv; loops over 4 N-tiles; K = 64 -> 2 WMMA per tile
    {
        const int mt = wv;
        for (int nt = 0; nt < 4; ++nt) {
            v8f acc = {0.f, 0.f, 0.f, 0.f, 0.f, 0.f, 0.f, 0.f};
            #pragma unroll
            for (int kb = 0; kb < 2; ++kb) {
                const int mrow = mt * 16 + ln16;
                v8h lo = *(const v8h*)&sH1[mrow][kb * 32 + g * 8];
                v8h hi = *(const v8h*)&sH1[mrow][kb * 32 + 16 + g * 8];
                v16h a;
                #pragma unroll
                for (int u = 0; u < 8; ++u) { a[u] = lo[u]; a[8 + u] = hi[u]; }
                v16h b = *(const v16h*)&W2p[(((size_t)kb * 4 + nt) * 32 + lane) * 16];
                acc = __builtin_amdgcn_wmma_f32_16x16x32_f16(false, a, false, b,
                                                             (short)0, acc, false, false);
            }
            const int ncol = nt * 16 + ln16;
            const float bias = br2[ncol];
            #pragma unroll
            for (int j = 0; j < 8; ++j)
                sH2[mt * 16 + j + 8 * g][ncol] = (_Float16)siluf(acc[j] + bias);
        }
    }
    __syncthreads();

    // ---------------- stage 2: w = h2 @ Wr3 + br3 (WMMA) + message scatter --
    // 8 M-tiles x 9 N-tiles = 72 tiles, 9 per wave (uniform).
    const float inv_s3 = 0.5773502691896258f;   // 1/sqrt(3)
    const float inv_s5 = 0.4472135954999579f;   // 1/sqrt(5)
    for (int t3 = wv; t3 < 72; t3 += 8) {
        const int mt = t3 / 9, nt = t3 % 9;
        v8f acc = {0.f, 0.f, 0.f, 0.f, 0.f, 0.f, 0.f, 0.f};
        #pragma unroll
        for (int kb = 0; kb < 2; ++kb) {
            const int mrow = mt * 16 + ln16;
            v8h lo = *(const v8h*)&sH2[mrow][kb * 32 + g * 8];
            v8h hi = *(const v8h*)&sH2[mrow][kb * 32 + 16 + g * 8];
            v16h a;
            #pragma unroll
            for (int u = 0; u < 8; ++u) { a[u] = lo[u]; a[8 + u] = hi[u]; }
            v16h b = *(const v16h*)&W3p[(((size_t)kb * 9 + nt) * 32 + lane) * 16];
            acc = __builtin_amdgcn_wmma_f32_16x16x32_f16(false, a, false, b,
                                                         (short)0, acc, false, false);
        }
        const int ncol = nt * 16 + ln16;      // radial channel 0..143
        const float bias = br3[ncol];
        #pragma unroll
        for (int j = 0; j < 8; ++j) {
            const int el = mt * 16 + j + 8 * g;
            const float w = acc[j] + bias;
            const int d = sDst[el];
            if (ncol < 32) {                              // w_ss: scalar->scalar
                atomicAdd(&a0[d * 56 + ncol], w * sSe[el][ncol]);
            } else if (ncol < 64) {                       // w_sv: scalar->vector
                const int c = ncol - 32;
                const float ws_ = w * sSe[el][c];
                atomicAdd(&a1[d * 144 + c * 3 + 0], ws_ * sY[el][0]);
                atomicAdd(&a1[d * 144 + c * 3 + 1], ws_ * sY[el][1]);
                atomicAdd(&a1[d * 144 + c * 3 + 2], ws_ * sY[el][2]);
            } else if (ncol < 96) {                       // w_sq: scalar->quad
                const int c = ncol - 64;
                const float ws_ = w * sSe[el][c];
                #pragma unroll
                for (int m = 0; m < 5; ++m)
                    atomicAdd(&a2[d * 200 + c * 5 + m], ws_ * sY[el][3 + m]);
            } else if (ncol < 112) {                      // w_vs: vector->vector
                const int c = ncol - 96;
                #pragma unroll
                for (int m = 0; m < 3; ++m)
                    atomicAdd(&a1[d * 144 + (32 + c) * 3 + m], w * sVe[el][c * 3 + m]);
            } else if (ncol < 128) {                      // w_vv: vector.Y1 -> scalar
                const int c = ncol - 112;
                float dv = sVe[el][c * 3 + 0] * sY[el][0]
                         + sVe[el][c * 3 + 1] * sY[el][1]
                         + sVe[el][c * 3 + 2] * sY[el][2];
                atomicAdd(&a0[d * 56 + 32 + c], w * dv * inv_s3);
            } else if (ncol < 136) {                      // w_qs: quad->quad
                const int c = ncol - 128;
                #pragma unroll
                for (int m = 0; m < 5; ++m)
                    atomicAdd(&a2[d * 200 + (32 + c) * 5 + m], w * sQe[el][c * 5 + m]);
            } else {                                      // w_qq: quad.Y2 -> scalar
                const int c = ncol - 136;
                float dq = 0.0f;
                #pragma unroll
                for (int m = 0; m < 5; ++m) dq += sQe[el][c * 5 + m] * sY[el][3 + m];
                atomicAdd(&a0[d * 56 + 48 + c], w * dq * inv_s5);
            }
        }
    }
}

// ---------------------------------------------------------------------------
// Node update (one layer): one block (64 thr) per node.
// ---------------------------------------------------------------------------
__global__ __launch_bounds__(64)
void rna_node_kernel(const float* __restrict__ a0, const float* __restrict__ a1,
                     const float* __restrict__ a2,
                     const float* __restrict__ Wm0, const float* __restrict__ Wm1,
                     const float* __restrict__ Wm2,
                     const float* __restrict__ Ws0, const float* __restrict__ Ws1,
                     const float* __restrict__ Ws2,
                     const float* __restrict__ Wg,  const float* __restrict__ bg,
                     float* __restrict__ s, float* __restrict__ v, float* __restrict__ q) {
    __shared__ float a0l[56], sl[32], a1l[144], vl[48], a2l[200], ql[40], nsl[32], gl[24];
    const int n = blockIdx.x, t = threadIdx.x;
    for (int i = t; i < 56;  i += 64) a0l[i] = a0[n * 56 + i];
    for (int i = t; i < 32;  i += 64) sl[i]  = s[n * 32 + i];
    for (int i = t; i < 144; i += 64) a1l[i] = a1[n * 144 + i];
    for (int i = t; i < 48;  i += 64) vl[i]  = v[n * 48 + i];
    for (int i = t; i < 200; i += 64) a2l[i] = a2[n * 200 + i];
    for (int i = t; i < 40;  i += 64) ql[i]  = q[n * 40 + i];
    __syncthreads();
    const float inv32 = 1.0f / 32.0f;        // 1/AVG_NEI (folded post-sum, linear)
    if (t < 32) {
        float acc = 0.0f;
        for (int c = 0; c < 56; ++c) acc += a0l[c] * Wm0[c * 32 + t];
        acc *= inv32;
        for (int c = 0; c < 32; ++c) acc += sl[c] * Ws0[c * 32 + t];
        nsl[t] = acc;
    }
    __syncthreads();
    if (t < 24) {
        float acc = bg[t];
        for (int c = 0; c < 32; ++c) acc += nsl[c] * Wg[c * 24 + t];
        gl[t] = sigmf(acc);
    }
    __syncthreads();
    if (t < 32) s[n * 32 + t] = sl[t] + siluf(nsl[t]);
    if (t < 48) {
        const int o = t / 3, m = t % 3;
        float acc = 0.0f;
        for (int c = 0; c < 48; ++c) acc += a1l[c * 3 + m] * Wm1[c * 16 + o];
        acc *= inv32;
        for (int c = 0; c < 16; ++c) acc += vl[c * 3 + m] * Ws1[c * 16 + o];
        v[n * 48 + t] = vl[t] + acc * gl[o];
    }
    if (t < 40) {
        const int o = t / 5, m = t % 5;
        float acc = 0.0f;
        for (int c = 0; c < 40; ++c) acc += a2l[c * 5 + m] * Wm2[c * 8 + o];
        acc *= inv32;
        for (int c = 0; c < 8; ++c) acc += ql[c * 5 + m] * Ws2[c * 8 + o];
        q[n * 40 + t] = ql[t] + acc * gl[16 + o];
    }
}

// ---------------------------------------------------------------------------
// Pooling MLP per node: logits = silu(silu(s@Wp1+b)@Wp2+b)@Wp3+b
// ---------------------------------------------------------------------------
__global__ __launch_bounds__(128)
void rna_pool_mlp_kernel(const float* __restrict__ s,
                         const float* __restrict__ Wp1, const float* __restrict__ bp1,
                         const float* __restrict__ Wp2, const float* __restrict__ bp2,
                         const float* __restrict__ Wp3, const float* __restrict__ bp3,
                         float* __restrict__ logits) {
    __shared__ float h1[128], h2[128], red[128];
    const int n = blockIdx.x, t = threadIdx.x;
    float acc = bp1[t];
    for (int c = 0; c < 32; ++c) acc += s[n * 32 + c] * Wp1[c * 128 + t];
    h1[t] = siluf(acc);
    __syncthreads();
    acc = bp2[t];
    for (int c = 0; c < 128; ++c) acc += h1[c] * Wp2[c * 128 + t];
    h2[t] = siluf(acc);
    __syncthreads();
    red[t] = h2[t] * Wp3[t];
    __syncthreads();
    for (int off = 64; off > 0; off >>= 1) {
        if (t < off) red[t] += red[t + off];
        __syncthreads();
    }
    if (t == 0) logits[n] = red[0] + bp3[0];
}

// ---------------------------------------------------------------------------
// Attention pooling over batch (B=8): softmax(logits per graph), gpool = sum s*att
// Single block (work is only 5000x32).
// ---------------------------------------------------------------------------
__global__ __launch_bounds__(256)
void rna_attn_pool_kernel(const float* __restrict__ logits, const int* __restrict__ batch,
                          const float* __restrict__ s, float* __restrict__ gpool) {
    __shared__ float red[256 * 8];
    __shared__ float smx[8], srz[8];
    __shared__ float gp[256];
    const int t = threadIdx.x;
    float lm[8];
    #pragma unroll
    for (int b = 0; b < 8; ++b) lm[b] = -1e30f;
    for (int n = t; n < NN; n += 256) {
        int b = batch[n];
        lm[b] = fmaxf(lm[b], logits[n]);
    }
    #pragma unroll
    for (int b = 0; b < 8; ++b) red[t * 8 + b] = lm[b];
    __syncthreads();
    if (t < 8) {
        float m = -1e30f;
        for (int k = 0; k < 256; ++k) m = fmaxf(m, red[k * 8 + t]);
        smx[t] = m;
    }
    __syncthreads();
    float ls[8];
    #pragma unroll
    for (int b = 0; b < 8; ++b) ls[b] = 0.0f;
    for (int n = t; n < NN; n += 256) {
        int b = batch[n];
        ls[b] += __expf(logits[n] - smx[b]);
    }
    #pragma unroll
    for (int b = 0; b < 8; ++b) red[t * 8 + b] = ls[b];
    __syncthreads();
    if (t < 8) {
        float sm = 0.0f;
        for (int k = 0; k < 256; ++k) sm += red[k * 8 + t];
        srz[t] = 1.0f / sm;   // exact divide once per graph
    }
    gp[t] = 0.0f;
    __syncthreads();
    for (int n = t; n < NN; n += 256) {
        int b = batch[n];
        float att = __expf(logits[n] - smx[b]) * srz[b];
        for (int c = 0; c < 32; ++c) atomicAdd(&gp[b * 32 + c], s[n * 32 + c] * att);
    }
    __syncthreads();
    gpool[t] = gp[t];   // 8*32 == 256 exactly
}

// ---------------------------------------------------------------------------
// Output head: ln(silu(ln(gpool@Wo1+bo1))@Wo2+bo2). One block per graph.
// ---------------------------------------------------------------------------
__global__ __launch_bounds__(256)
void rna_out_kernel(const float* __restrict__ gpool,
                    const float* __restrict__ Wo1, const float* __restrict__ bo1,
                    const float* __restrict__ g1,  const float* __restrict__ be1,
                    const float* __restrict__ Wo2, const float* __restrict__ bo2,
                    const float* __restrict__ g2,  const float* __restrict__ be2,
                    float* __restrict__ out) {
    __shared__ float g32[32];
    __shared__ float hb[1024];
    __shared__ float red[256];
    __shared__ float mv[2];
    const int b = blockIdx.x, t = threadIdx.x;
    if (t < 32) g32[t] = gpool[b * 32 + t];
    __syncthreads();
    float loc[4];
    #pragma unroll
    for (int i = 0; i < 4; ++i) {
        const int k = t + i * 256;
        float acc = bo1[k];
        for (int c = 0; c < 32; ++c) acc += g32[c] * Wo1[c * 1024 + k];
        loc[i] = acc;
    }
    // LayerNorm over 1024
    red[t] = loc[0] + loc[1] + loc[2] + loc[3];
    __syncthreads();
    for (int off = 128; off > 0; off >>= 1) { if (t < off) red[t] += red[t + off]; __syncthreads(); }
    if (t == 0) mv[0] = red[0] * (1.0f / 1024.0f);
    __syncthreads();
    float m = mv[0];
    float ss = 0.0f;
    #pragma unroll
    for (int i = 0; i < 4; ++i) { float d = loc[i] - m; ss += d * d; }
    red[t] = ss;
    __syncthreads();
    for (int off = 128; off > 0; off >>= 1) { if (t < off) red[t] += red[t + off]; __syncthreads(); }
    if (t == 0) mv[1] = red[0] * (1.0f / 1024.0f);
    __syncthreads();
    float rstd = rsqrtf(mv[1] + 1e-5f);
    #pragma unroll
    for (int i = 0; i < 4; ++i) {
        const int k = t + i * 256;
        hb[k] = siluf((loc[i] - m) * rstd * g1[k] + be1[k]);
    }
    __syncthreads();
    float lo2[2];
    #pragma unroll
    for (int i = 0; i < 2; ++i) {
        const int o = t + i * 256;
        float acc = bo2[o];
        for (int c = 0; c < 1024; ++c) acc += hb[c] * Wo2[c * 512 + o];
        lo2[i] = acc;
    }
    // LayerNorm over 512
    red[t] = lo2[0] + lo2[1];
    __syncthreads();
    for (int off = 128; off > 0; off >>= 1) { if (t < off) red[t] += red[t + off]; __syncthreads(); }
    if (t == 0) mv[0] = red[0] * (1.0f / 512.0f);
    __syncthreads();
    m = mv[0];
    ss = 0.0f;
    #pragma unroll
    for (int i = 0; i < 2; ++i) { float d = lo2[i] - m; ss += d * d; }
    red[t] = ss;
    __syncthreads();
    for (int off = 128; off > 0; off >>= 1) { if (t < off) red[t] += red[t + off]; __syncthreads(); }
    if (t == 0) mv[1] = red[0] * (1.0f / 512.0f);
    __syncthreads();
    rstd = rsqrtf(mv[1] + 1e-5f);
    #pragma unroll
    for (int i = 0; i < 2; ++i) {
        const int o = t + i * 256;
        out[b * 512 + o] = (lo2[i] - m) * rstd * g2[o] + be2[o];
    }
}

// ---------------------------------------------------------------------------
// Host launch
// ---------------------------------------------------------------------------
extern "C" void kernel_launch(void* const* d_in, const int* in_sizes, int n_in,
                              void* d_out, int out_size, void* d_ws, size_t ws_size,
                              hipStream_t stream) {
    (void)in_sizes; (void)n_in; (void)out_size; (void)ws_size;

    const float* x    = (const float*)d_in[0];
    const float* pos  = (const float*)d_in[1];
    const float* Win  = (const float*)d_in[2];
    const float* Wr1  = (const float*)d_in[3];
    const float* br1  = (const float*)d_in[4];
    const float* Wr2  = (const float*)d_in[5];
    const float* br2  = (const float*)d_in[6];
    const float* Wr3  = (const float*)d_in[7];
    const float* br3  = (const float*)d_in[8];
    const float* Wm0  = (const float*)d_in[9];
    const float* Wm1  = (const float*)d_in[10];
    const float* Wm2  = (const float*)d_in[11];
    const float* Ws0  = (const float*)d_in[12];
    const float* Ws1  = (const float*)d_in[13];
    const float* Ws2  = (const float*)d_in[14];
    const float* Wg   = (const float*)d_in[15];
    const float* bg   = (const float*)d_in[16];
    const float* Wp1  = (const float*)d_in[17];
    const float* bp1  = (const float*)d_in[18];
    const float* Wp2  = (const float*)d_in[19];
    const float* bp2  = (const float*)d_in[20];
    const float* Wp3  = (const float*)d_in[21];
    const float* bp3  = (const float*)d_in[22];
    const float* Wo1  = (const float*)d_in[23];
    const float* bo1  = (const float*)d_in[24];
    const float* g1   = (const float*)d_in[25];
    const float* be1  = (const float*)d_in[26];
    const float* Wo2  = (const float*)d_in[27];
    const float* bo2  = (const float*)d_in[28];
    const float* g2   = (const float*)d_in[29];
    const float* be2  = (const float*)d_in[30];
    const int*   ei   = (const int*)d_in[31];
    const int*   bat  = (const int*)d_in[32];
    float* out = (float*)d_out;

    float* W = (float*)d_ws;
    float* embW = W + OFF_EMB;
    float* YW   = W + OFF_Y;
    float* sW   = W + OFF_S;
    float* vW   = W + OFF_V;
    float* qW   = W + OFF_Q;
    float* a0W  = W + OFF_A0;
    float* a1W  = W + OFF_A1;
    float* a2W  = W + OFF_A2;
    float* logW = W + OFF_LOG;
    float* gpW  = W + OFF_GP;
    _Float16* W2p = (_Float16*)(W + OFF_PACK);
    _Float16* W3p = W2p + (size_t)NLAYER * 4096;

    // one-time per launch: pack radial weights to f16 WMMA layout
    rna_pack_wr2_kernel<<<(NLAYER * 4096 + 255) / 256, 256, 0, stream>>>(Wr2, W2p);
    rna_pack_wr3_kernel<<<(NLAYER * 9216 + 255) / 256, 256, 0, stream>>>(Wr3, W3p);

    // edge geometry (reused across layers)
    rna_geom_kernel<<<(NE + 255) / 256, 256, 0, stream>>>(pos, ei, embW, YW);

    // initial node state
    rna_init_s_kernel<<<(NN * MUL0 + 255) / 256, 256, 0, stream>>>(x, Win, sW);
    rna_zero_kernel<<<((NN * 48 + NN * 40) + 255) / 256, 256, 0, stream>>>(vW, NN * 48 + NN * 40);

    const int aTotal = NN * 56 + NN * 144 + NN * 200; // a0|a1|a2 contiguous
    for (int i = 0; i < NLAYER; ++i) {
        rna_zero_kernel<<<(aTotal + 255) / 256, 256, 0, stream>>>(a0W, aTotal);
        rna_edge_kernel<<<NE / 128, 256, 0, stream>>>(
            ei, embW, YW, sW, vW, qW,
            Wr1 + (size_t)i * 512, br1 + (size_t)i * 64,
            W2p + (size_t)i * 4096, br2 + (size_t)i * 64,
            W3p + (size_t)i * 9216, br3 + (size_t)i * 144,
            a0W, a1W, a2W);
        rna_node_kernel<<<NN, 64, 0, stream>>>(
            a0W, a1W, a2W,
            Wm0 + (size_t)i * 1792, Wm1 + (size_t)i * 768, Wm2 + (size_t)i * 320,
            Ws0 + (size_t)i * 1024, Ws1 + (size_t)i * 256, Ws2 + (size_t)i * 64,
            Wg + (size_t)i * 768, bg + (size_t)i * 24,
            sW, vW, qW);
    }

    rna_pool_mlp_kernel<<<NN, 128, 0, stream>>>(sW, Wp1, bp1, Wp2, bp2, Wp3, bp3, logW);
    rna_attn_pool_kernel<<<1, 256, 0, stream>>>(logW, bat, sW, gpW);
    rna_out_kernel<<<NBATCH, 256, 0, stream>>>(gpW, Wo1, bo1, g1, be1, Wo2, bo2, g2, be2, out);
}
